// LoraLinear_22548578304347
// MI455X (gfx1250) — compile-verified, matrix operand order checked by reading, not run on hardware
//
#include <hip/hip_runtime.h>

typedef __attribute__((ext_vector_type(16))) __bf16 v16bf;
typedef __attribute__((ext_vector_type(8)))  float  v8f;
typedef __attribute__((ext_vector_type(4)))  float  v4f;
typedef __attribute__((ext_vector_type(4)))  __bf16 v4bf;
typedef __attribute__((ext_vector_type(4)))  unsigned int u32x4;
typedef __attribute__((ext_vector_type(8)))  int i32x8;
typedef __attribute__((ext_vector_type(4)))  int i32x4;

#define IN_F   4096
#define OUT_F  4096
#define M_TOT  8192
#define SCALING 2.0f          // ALPHA / RANK = 32 / 16
#define LDS_STRIDE_H 40       // 32 data halves + 4-DWORD pad per row (80 B) == TDM pad fields
#define TILE_H (128 * LDS_STRIDE_H)

#define HAVE_TDM __has_builtin(__builtin_amdgcn_tensor_load_to_lds)

#if __has_builtin(__builtin_amdgcn_s_wait_tensorcnt)
#define WAIT_TENSORCNT(n) __builtin_amdgcn_s_wait_tensorcnt(n)
#else
#define WAIT_TENSORCNT(n) asm volatile("s_wait_tensorcnt %0" ::"i"(n) : "memory")
#endif

// ---------------------------------------------------------------------------
// f32 -> bf16 bulk convert (vectorized x4, grid-stride)
// ---------------------------------------------------------------------------
__global__ __launch_bounds__(256) void cvt_bf16_kernel(const float* __restrict__ src,
                                                       __bf16* __restrict__ dst,
                                                       size_t n4) {
    size_t i = (size_t)blockIdx.x * blockDim.x + threadIdx.x;
    size_t stride = (size_t)gridDim.x * blockDim.x;
    for (; i < n4; i += stride) {
        v4f f = ((const v4f*)src)[i];
        ((v4bf*)dst)[i] = __builtin_convertvector(f, v4bf);
    }
}

// ---------------------------------------------------------------------------
// xa[m, 0:16] = SCALING * (x[m,:] @ lora_A^T), stored bf16 zero-padded to 32
// ---------------------------------------------------------------------------
__global__ __launch_bounds__(256) void lora_xa_kernel(const float* __restrict__ x,
                                                      const float* __restrict__ lA,
                                                      __bf16* __restrict__ xab) {
    __shared__ float red[16][257];
    const int m = blockIdx.x;
    const int t = threadIdx.x;
    const float* xr = x + (size_t)m * IN_F;

    float acc[16];
#pragma unroll
    for (int r = 0; r < 16; ++r) acc[r] = 0.0f;

    const int k0 = t * 16;
#pragma unroll 4
    for (int i = 0; i < 16; ++i) {
        const float xv = xr[k0 + i];
#pragma unroll
        for (int r = 0; r < 16; ++r) acc[r] += xv * lA[r * IN_F + k0 + i];
    }
#pragma unroll
    for (int r = 0; r < 16; ++r) red[r][t] = acc[r];
    __syncthreads();

    if (t < 16) {
        float s = 0.0f;
        for (int j = 0; j < 256; ++j) s += red[t][j];
        xab[(size_t)m * 32 + t] = (__bf16)(s * SCALING);
    } else if (t < 32) {
        xab[(size_t)m * 32 + t] = (__bf16)0.0f;
    }
}

// ---------------------------------------------------------------------------
// Bb[n, 0:16] = bf16(lora_B[n, :]), zero-padded to 32
// ---------------------------------------------------------------------------
__global__ __launch_bounds__(256) void padB_kernel(const float* __restrict__ lB,
                                                   __bf16* __restrict__ Bb) {
    const int i = blockIdx.x * 256 + threadIdx.x;
    const int n = i >> 5, r = i & 31;
    Bb[i] = (r < 16) ? (__bf16)lB[n * 16 + r] : (__bf16)0.0f;
}

// ---------------------------------------------------------------------------
// Main fused GEMM: out = xb @ Wb^T + bias + xab @ Bb^T
// 128x128 tile / workgroup, 8 waves; TDM-staged double-buffered LDS
// ---------------------------------------------------------------------------
__global__ __launch_bounds__(256) void lora_gemm_wmma(
    const __bf16* __restrict__ xb, const __bf16* __restrict__ Wb,
    const __bf16* __restrict__ xab, const __bf16* __restrict__ Bb,
    const float* __restrict__ bias, float* __restrict__ out) {

    __shared__ __bf16 As[2][TILE_H];
    __shared__ __bf16 Bs[2][TILE_H];

    const int t    = threadIdx.x;
    const int lane = t & 31;
    const int wave = t >> 5;
    const int wm   = wave & 3;            // M sub-tile 0..3  (*32 rows)
    const int wn   = wave >> 2;           // N sub-tile 0..1  (*64 cols)
    const int m0   = blockIdx.y * 128;
    const int n0   = blockIdx.x * 128;

    // fragment load per documented 16-bit A/B layout:
    // lane m = lane&15; lane[4] selects koff (0 or 8 halves);
    // halves 0..7 = K[koff..koff+7], halves 8..15 = K[16+koff..16+koff+7]
    auto ldfrag = [&](const __bf16* lds, int r0) -> v16bf {
        union { v16bf v; uint4 q[2]; } f;
        const char* p = (const char*)lds +
                        (r0 + (lane & 15)) * (LDS_STRIDE_H * 2) + ((lane >> 4) << 4);
        f.q[0] = *(const uint4*)p;
        f.q[1] = *(const uint4*)(p + 32);
        return f.v;
    };

    v8f acc[2][4];
    const v8f vzero = {0.f, 0.f, 0.f, 0.f, 0.f, 0.f, 0.f, 0.f};
#pragma unroll
    for (int i = 0; i < 2; ++i)
#pragma unroll
        for (int j = 0; j < 4; ++j) acc[i][j] = vzero;

    auto mma_step = [&](const __bf16* Atile, const __bf16* Btile) {
        v16bf a0 = ldfrag(Atile, wm * 32);
        v16bf a1 = ldfrag(Atile, wm * 32 + 16);
        v16bf bf[4];
#pragma unroll
        for (int nf = 0; nf < 4; ++nf) bf[nf] = ldfrag(Btile, wn * 64 + nf * 16);
#pragma unroll
        for (int nf = 0; nf < 4; ++nf)
            acc[0][nf] = __builtin_amdgcn_wmma_f32_16x16x32_bf16(
                false, a0, false, bf[nf], (short)0, acc[0][nf], false, false);
#pragma unroll
        for (int nf = 0; nf < 4; ++nf)
            acc[1][nf] = __builtin_amdgcn_wmma_f32_16x16x32_bf16(
                false, a1, false, bf[nf], (short)0, acc[1][nf], false, false);
    };

#if HAVE_TDM
    // ---- Tensor Data Mover staging: one DMA per 128x32 bf16 tile ----
    // D# per CDNA5 ISA ch.8: pad_interval=3 (every 16 DWORDs = 64B row),
    // pad_amount=3 (+4 DWORDs) -> 80B LDS row stride == LDS_STRIDE_H.
    const unsigned ldsA0 = (unsigned)(uintptr_t)&As[0][0];
    const unsigned ldsA1 = (unsigned)(uintptr_t)&As[1][0];
    const unsigned ldsB0 = (unsigned)(uintptr_t)&Bs[0][0];
    const unsigned ldsB1 = (unsigned)(uintptr_t)&Bs[1][0];

    auto tdm_issue = [&](const __bf16* gp, unsigned lds_u, int stride_e) {
        const unsigned long long ga = (unsigned long long)(uintptr_t)gp;
        u32x4 g0;
        g0[0] = 1u;                                        // count=1, user mode
        g0[1] = lds_u;                                     // lds_addr
        g0[2] = (unsigned)(ga & 0xFFFFFFFFu);              // global_addr[31:0]
        g0[3] = (unsigned)((ga >> 32) & 0x1FFFFFFu) | (2u << 30);  // addr hi | type=2
        i32x8 g1;
        g1[0] = (1 << 16) | (1 << 20) | (3 << 22) | (3 << 25); // 2B elems, pad en, ivl, amt
        g1[1] = (int)(((unsigned)stride_e & 0xFFFFu) << 16);   // tensor_dim0 lo16
        g1[2] = (int)((((unsigned)stride_e >> 16) & 0xFFFFu) | (8192u << 16)); // d0 hi, d1 lo
        g1[3] = (int)(32u << 16);                              // d1 hi=0 | tile_dim0=32
        g1[4] = (int)128u;                                     // tile_dim1=128, tile_dim2=0
        g1[5] = stride_e;                                      // tensor_dim0_stride lo32
        g1[6] = 0;                                             // stride hi | d1_stride lo
        g1[7] = 0;
        const i32x4 gz4 = {0, 0, 0, 0};
        const i32x8 gz8 = {0, 0, 0, 0, 0, 0, 0, 0};
        // clang-23 / therock-10.0 lane: 6-arg form (extra i32x8 before cpol)
        __builtin_amdgcn_tensor_load_to_lds(g0, g1, gz4, gz4, gz8, 0);
    };

    auto issue_step = [&](int kt, int buf) {               // wave-uniform, wave 0 only
        if (kt < IN_F / 32) {
            tdm_issue(xb + (size_t)m0 * IN_F + kt * 32, buf ? ldsA1 : ldsA0, IN_F);
            tdm_issue(Wb + (size_t)n0 * IN_F + kt * 32, buf ? ldsB1 : ldsB0, IN_F);
        } else {                                           // LoRA rank-16 tail tiles
            tdm_issue(xab + (size_t)m0 * 32, buf ? ldsA1 : ldsA0, 32);
            tdm_issue(Bb + (size_t)n0 * 32, buf ? ldsB1 : ldsB0, 32);
        }
    };

    if (wave == 0) issue_step(0, 0);
    for (int kt = 0; kt <= IN_F / 32; ++kt) {              // 128 base steps + LoRA tail
        const int cur = kt & 1;
        if (kt < IN_F / 32) {
            if (wave == 0) issue_step(kt + 1, cur ^ 1);    // prefetch next tile via TDM
            WAIT_TENSORCNT(2);                             // current tile landed in LDS
        } else {
            WAIT_TENSORCNT(0);
        }
        __syncthreads();                                   // publish LDS to all waves
        mma_step(As[cur], Bs[cur]);
        __syncthreads();                                   // done reading before overwrite
    }
#else
    // ---- fallback: synchronous per-lane staging (round-1 path) ----
    const int srow = t >> 1;
    const int scb  = (t & 1) << 5;
    auto stage = [&](const __bf16* src, int row0, size_t stride_h, int k0_h, __bf16* lds) {
        const char* g = (const char*)(src + (size_t)(row0 + srow) * stride_h + k0_h) + scb;
        uint4 q0 = *(const uint4*)g;
        uint4 q1 = *(const uint4*)(g + 16);
        char* l = (char*)lds + srow * (LDS_STRIDE_H * 2) + scb;
        *(uint4*)l        = q0;
        *(uint4*)(l + 16) = q1;
    };
    for (int kt = 0; kt < IN_F / 32; ++kt) {
        stage(xb, m0, IN_F, kt * 32, As[0]);
        stage(Wb, n0, IN_F, kt * 32, Bs[0]);
        __syncthreads();
        mma_step(As[0], Bs[0]);
        __syncthreads();
    }
    stage(xab, m0, 32, 0, As[0]);
    stage(Bb,  n0, 32, 0, Bs[0]);
    __syncthreads();
    mma_step(As[0], Bs[0]);
#endif

    // ---- epilogue: bias add + store (C/D layout: vgpr r -> row, lane -> col) ----
    const int lr = lane & 15;
    const int lh = (lane >> 4) * 8;
#pragma unroll
    for (int mf = 0; mf < 2; ++mf) {
        const int row = m0 + wm * 32 + mf * 16 + lh;
#pragma unroll
        for (int nf = 0; nf < 4; ++nf) {
            const int col = n0 + wn * 64 + nf * 16 + lr;
            const float bv = bias[col];
#pragma unroll
            for (int r = 0; r < 8; ++r)
                out[(size_t)(row + r) * OUT_F + col] = acc[mf][nf][r] + bv;
        }
    }
}

// ---------------------------------------------------------------------------
extern "C" void kernel_launch(void* const* d_in, const int* in_sizes, int n_in,
                              void* d_out, int out_size, void* d_ws, size_t ws_size,
                              hipStream_t stream) {
    const float* x  = (const float*)d_in[0];   // [4,2048,4096]
    const float* W  = (const float*)d_in[1];   // [4096,4096]
    const float* b  = (const float*)d_in[2];   // [4096]
    const float* lA = (const float*)d_in[3];   // [16,4096]
    const float* lB = (const float*)d_in[4];   // [4096,16]
    float* out = (float*)d_out;

    // workspace layout (bf16): xb | Wb | xab | Bb  (~97 MB)
    __bf16* xb  = (__bf16*)d_ws;
    __bf16* Wb  = xb  + (size_t)M_TOT * IN_F;
    __bf16* xab = Wb  + (size_t)OUT_F * IN_F;
    __bf16* Bb  = xab + (size_t)M_TOT * 32;

    cvt_bf16_kernel<<<4096, 256, 0, stream>>>(x, xb, (size_t)M_TOT * IN_F / 4);
    cvt_bf16_kernel<<<4096, 256, 0, stream>>>(W, Wb, (size_t)OUT_F * IN_F / 4);
    lora_xa_kernel<<<M_TOT, 256, 0, stream>>>(x, lA, xab);
    padB_kernel<<<(OUT_F * 32) / 256, 256, 0, stream>>>(lB, Bb);

    dim3 grid(OUT_F / 128, M_TOT / 128);   // 32 x 64 workgroups
    lora_gemm_wmma<<<grid, 256, 0, stream>>>(xb, Wb, xab, Bb, b, out);
}
